// AutoCorrelation_45561013076193
// MI455X (gfx1250) — matrix operand order repaired and checked by reading, not compiled, not run on hardware
//
#include <hip/hip_runtime.h>

// Problem constants (from reference): B=16, C=512, T=2048, K=3
#define Bn 16
#define Cn 512
#define Tn 2048
#define Kn 3

typedef __attribute__((ext_vector_type(16))) _Float16 v16h;
typedef __attribute__((ext_vector_type(8)))  _Float16 v8h;
typedef __attribute__((ext_vector_type(2)))  __fp16   h2;   // matches __builtin_amdgcn_cvt_pkrtz return type
typedef __attribute__((ext_vector_type(8)))  float    v8f;

__device__ __forceinline__ v16h cat16(v8h lo, v8h hi) {
    return __builtin_shufflevector(lo, hi, 0, 1, 2, 3, 4, 5, 6, 7,
                                           8, 9, 10, 11, 12, 13, 14, 15);
}

// LDS tile row stride in halves: 32 payload + 8 pad = 80B rows (16B multiple,
// keeps every 8-half chunk 16B aligned for ds_load_b128, breaks bank conflicts)
#define LDK 40

// ---------------------------------------------------------------------------
// WMMA GEMM for the q/k/v projections, LDS-staged.
//   Out[b,m,n] = sum_k A[k*M + m] * Bsrc[b,k,n] + bias[m]
// Block: 256 threads = 8 waves. Block tile 128(M) x 64(N), K step 32.
// Tiles converted f32->f16 on the way into LDS (packed cvt + b32 ds stores),
// fragments assembled from LDS with two ds_load_b128 per operand.
// ---------------------------------------------------------------------------
__global__ void __launch_bounds__(256) proj_gemm_kernel(
    const float* __restrict__ A, const float* __restrict__ Bsrc,
    const float* __restrict__ bias, float* __restrict__ Out,
    int M, int N, int Kdim)
{
    __shared__ __align__(16) _Float16 lA[128][LDK];  // [m][k]  k-contiguous
    __shared__ __align__(16) _Float16 lB[64][LDK];   // [n][k]  k-contiguous

    const int tid  = threadIdx.x;
    const int lane = tid & 31;
    const int wave = tid >> 5;
    const int b    = blockIdx.z;
    const int n0   = blockIdx.x * 64;
    const int m0b  = blockIdx.y * 128;
    const int lhi  = (lane >> 4) & 1;
    const int l15  = lane & 15;

    v8f acc[4] = {};
    const float* Bb = Bsrc + (size_t)b * Kdim * N;

    for (int k0 = 0; k0 < Kdim; k0 += 32) {
        __syncthreads();
        // ---- stage A tile: global [k][m] (m contig) -> LDS [m][k] (k contig)
        // 512 tasks: (k-pair, 4-wide m group); float4 loads, packed f16 stores.
        for (int i = tid; i < 512; i += 256) {
            const int kp = i >> 5;          // 0..15  -> k = 2*kp
            const int mg = i & 31;          // 0..31  -> m = 4*mg
            const int k  = kp * 2;
            const float4 a0 = *(const float4*)&A[(size_t)(k0 + k)     * M + m0b + mg * 4];
            const float4 a1 = *(const float4*)&A[(size_t)(k0 + k + 1) * M + m0b + mg * 4];
            *(h2*)&lA[mg * 4 + 0][k] = __builtin_amdgcn_cvt_pkrtz(a0.x, a1.x);
            *(h2*)&lA[mg * 4 + 1][k] = __builtin_amdgcn_cvt_pkrtz(a0.y, a1.y);
            *(h2*)&lA[mg * 4 + 2][k] = __builtin_amdgcn_cvt_pkrtz(a0.z, a1.z);
            *(h2*)&lA[mg * 4 + 3][k] = __builtin_amdgcn_cvt_pkrtz(a0.w, a1.w);
        }
        // ---- stage B tile: global [k][n] (n contig) -> LDS [n][k] (k contig)
        {
            const int kp = tid >> 4;        // 0..15
            const int ng = tid & 15;        // 0..15
            const int k  = kp * 2;
            const float4 b0 = *(const float4*)&Bb[(size_t)(k0 + k)     * N + n0 + ng * 4];
            const float4 b1 = *(const float4*)&Bb[(size_t)(k0 + k + 1) * N + n0 + ng * 4];
            *(h2*)&lB[ng * 4 + 0][k] = __builtin_amdgcn_cvt_pkrtz(b0.x, b1.x);
            *(h2*)&lB[ng * 4 + 1][k] = __builtin_amdgcn_cvt_pkrtz(b0.y, b1.y);
            *(h2*)&lB[ng * 4 + 2][k] = __builtin_amdgcn_cvt_pkrtz(b0.z, b1.z);
            *(h2*)&lB[ng * 4 + 3][k] = __builtin_amdgcn_cvt_pkrtz(b0.w, b1.w);
        }
        if (k0 + 32 < Kdim)
            __builtin_prefetch(&Bb[(size_t)(k0 + 32) * N + n0 + lane], 0, 1);
        __syncthreads();

        // ---- fragments from LDS
        // A(m,k): elem e -> k = (e&7) + ((e&8)<<1) + 8*laneHi  => chunks at 8*lhi, 16+8*lhi
        const int ml = wave * 16 + l15;
        const v8h a0 = *(const v8h*)&lA[ml][8 * lhi];
        const v8h a1 = *(const v8h*)&lA[ml][16 + 8 * lhi];
        const v16h afrag = cat16(a0, a1);
#pragma unroll
        for (int nn = 0; nn < 4; ++nn) {
            // B(k,n): elem e -> k = (e&7) + ((e&8)<<1) + 8*laneHi => chunks at 8*lhi, 16+8*lhi
            const int nl = nn * 16 + l15;
            const v8h b0 = *(const v8h*)&lB[nl][8 * lhi];
            const v8h b1 = *(const v8h*)&lB[nl][16 + 8 * lhi];
            acc[nn] = __builtin_amdgcn_wmma_f32_16x16x32_f16(
                false, afrag, false, cat16(b0, b1), (short)0, acc[nn], false, false);
        }
    }
    // D layout: vgpr r -> M = m0 + r + (laneHi<<3), N = lane&15
    const int m0 = m0b + wave * 16;
#pragma unroll
    for (int nn = 0; nn < 4; ++nn) {
#pragma unroll
        for (int r = 0; r < 8; ++r) {
            const int m = m0 + r + (lhi << 3);
            const int n = n0 + nn * 16 + l15;
            Out[((size_t)b * M + m) * N + n] = acc[nn][r] + bias[m];
        }
    }
}

// ---------------------------------------------------------------------------
// Fused per-row kernel: one block per (b,c) row.
//   z = q + i*k ; Z = FFT(z) (Stockham, ping-pong in LDS, no bit reversal)
//   S[f] = Q[f]*conj(K[f]) extracted from Z ; ac = IFFT(S) * 1/T^2
//   then softmax stats (max, sum exp) + top-3 lags, emit lags + weights only.
// ---------------------------------------------------------------------------
__global__ void __launch_bounds__(256) autocorr_topk_kernel(
    const float* __restrict__ Q, const float* __restrict__ Kmat,
    int* __restrict__ lags, float* __restrict__ wvals)
{
    __shared__ float2 bufA[Tn];
    __shared__ float2 bufB[Tn];
    __shared__ float  red[256];
    __shared__ float  cv[768];
    __shared__ int    ci[768];

    const int tid = threadIdx.x;
    const int row = blockIdx.x;                 // b*Cn + c
    const float* qr = Q    + (size_t)row * Tn;
    const float* kr = Kmat + (size_t)row * Tn;

    for (int i = tid; i < Tn; i += 256) { bufA[i].x = qr[i]; bufA[i].y = kr[i]; }
    __syncthreads();

    float2* src = bufA;
    float2* dst = bufB;

    // forward FFT (sign -1)
    for (int Ns = 1; Ns < Tn; Ns <<= 1) {
        for (int j = tid; j < Tn / 2; j += 256) {
            float2 v0 = src[j];
            float2 v1 = src[j + Tn / 2];
            const float ang = -6.283185307179586f * (float)(j % Ns) / (float)(Ns * 2);
            float sa, ca; __sincosf(ang, &sa, &ca);
            float2 w1; w1.x = v1.x * ca - v1.y * sa; w1.y = v1.x * sa + v1.y * ca;
            const int idxD = (j / Ns) * (Ns * 2) + (j % Ns);
            dst[idxD].x      = v0.x + w1.x; dst[idxD].y      = v0.y + w1.y;
            dst[idxD + Ns].x = v0.x - w1.x; dst[idxD + Ns].y = v0.y - w1.y;
        }
        float2* t = src; src = dst; dst = t;
        __syncthreads();
    }

    // S = Q * conj(K): Q=(Z[f]+conj(Z[-f]))/2, K=(Z[f]-conj(Z[-f]))/(2i)
    for (int f = tid; f < Tn; f += 256) {
        float2 zf = src[f];
        float2 zn = src[(Tn - f) & (Tn - 1)];
        const float Qr = 0.5f * (zf.x + zn.x);
        const float Qi = 0.5f * (zf.y - zn.y);
        const float Kr = 0.5f * (zf.y + zn.y);
        const float Ki = -0.5f * (zf.x - zn.x);
        dst[f].x = Qr * Kr + Qi * Ki;
        dst[f].y = Qi * Kr - Qr * Ki;
    }
    { float2* t = src; src = dst; dst = t; }
    __syncthreads();

    // inverse FFT (sign +1), unnormalized
    for (int Ns = 1; Ns < Tn; Ns <<= 1) {
        for (int j = tid; j < Tn / 2; j += 256) {
            float2 v0 = src[j];
            float2 v1 = src[j + Tn / 2];
            const float ang = 6.283185307179586f * (float)(j % Ns) / (float)(Ns * 2);
            float sa, ca; __sincosf(ang, &sa, &ca);
            float2 w1; w1.x = v1.x * ca - v1.y * sa; w1.y = v1.x * sa + v1.y * ca;
            const int idxD = (j / Ns) * (Ns * 2) + (j % Ns);
            dst[idxD].x      = v0.x + w1.x; dst[idxD].y      = v0.y + w1.y;
            dst[idxD + Ns].x = v0.x - w1.x; dst[idxD + Ns].y = v0.y - w1.y;
        }
        float2* t = src; src = dst; dst = t;
        __syncthreads();
    }

    const float scale = 1.0f / ((float)Tn * (float)Tn);

    // row max
    float mx = -3.402823466e38f;
    for (int t = tid; t < Tn; t += 256) mx = fmaxf(mx, src[t].x * scale);
    red[tid] = mx; __syncthreads();
    for (int s = 128; s > 0; s >>= 1) {
        if (tid < s) red[tid] = fmaxf(red[tid], red[tid + s]);
        __syncthreads();
    }
    const float rowmax = red[0];
    __syncthreads();

    // sum exp
    float sum = 0.f;
    for (int t = tid; t < Tn; t += 256) sum += __expf(src[t].x * scale - rowmax);
    red[tid] = sum; __syncthreads();
    for (int s = 128; s > 0; s >>= 1) {
        if (tid < s) red[tid] += red[tid + s];
        __syncthreads();
    }
    const float rowsum = red[0];

    // per-thread top-3
    float tv0 = -3.402823466e38f, tv1 = tv0, tv2 = tv0;
    int   ti0 = 0, ti1 = 0, ti2 = 0;
    for (int t = tid; t < Tn; t += 256) {
        const float v = src[t].x * scale;
        if (v > tv0)      { tv2 = tv1; ti2 = ti1; tv1 = tv0; ti1 = ti0; tv0 = v; ti0 = t; }
        else if (v > tv1) { tv2 = tv1; ti2 = ti1; tv1 = v; ti1 = t; }
        else if (v > tv2) { tv2 = v; ti2 = t; }
    }
    cv[tid * 3 + 0] = tv0; ci[tid * 3 + 0] = ti0;
    cv[tid * 3 + 1] = tv1; ci[tid * 3 + 1] = ti1;
    cv[tid * 3 + 2] = tv2; ci[tid * 3 + 2] = ti2;
    __syncthreads();

    if (tid == 0) {
        float bv0 = -3.402823466e38f, bv1 = bv0, bv2 = bv0;
        int   bi0 = 0, bi1 = 0, bi2 = 0;
        for (int i = 0; i < 768; ++i) {
            const float v = cv[i]; const int idx = ci[i];
            if (v > bv0)      { bv2 = bv1; bi2 = bi1; bv1 = bv0; bi1 = bi0; bv0 = v; bi0 = idx; }
            else if (v > bv1) { bv2 = bv1; bi2 = bi1; bv1 = v; bi1 = idx; }
            else if (v > bv2) { bv2 = v; bi2 = idx; }
        }
        lags[row * Kn + 0] = bi0; wvals[row * Kn + 0] = __expf(bv0 - rowmax) / rowsum;
        lags[row * Kn + 1] = bi1; wvals[row * Kn + 1] = __expf(bv1 - rowmax) / rowsum;
        lags[row * Kn + 2] = bi2; wvals[row * Kn + 2] = __expf(bv2 - rowmax) / rowsum;
    }
}

// ---------------------------------------------------------------------------
// Final projection GEMM, LDS-staged, with the roll/gather + weight scaling
// fused into the B-tile staging:
//   Out[b,d,t] = sum_{e=kk*C+c} Wf[e*C + d] * ( w[b,c,kk] * v[b,c,(t-lag)%T] ) + bf[d]
// Never materializes the [B, K*C, T] aggregate.
// ---------------------------------------------------------------------------
__global__ void __launch_bounds__(256) final_gemm_kernel(
    const float* __restrict__ Wf, const float* __restrict__ V,
    const int* __restrict__ lags, const float* __restrict__ wvals,
    const float* __restrict__ bias, float* __restrict__ Out)
{
    __shared__ __align__(16) _Float16 lA[128][LDK];  // [m][k]
    __shared__ __align__(16) _Float16 lB[64][LDK];   // [n][e]

    const int tid  = threadIdx.x;
    const int lane = tid & 31;
    const int wave = tid >> 5;
    const int b    = blockIdx.z;
    const int n0   = blockIdx.x * 64;
    const int m0b  = blockIdx.y * 128;
    const int lhi  = (lane >> 4) & 1;
    const int l15  = lane & 15;

    v8f acc[4] = {};

    for (int k0 = 0; k0 < Kn * Cn; k0 += 32) {
        __syncthreads();
        // ---- stage A tile (Wf): same pattern as projections
        for (int i = tid; i < 512; i += 256) {
            const int kp = i >> 5;
            const int mg = i & 31;
            const int k  = kp * 2;
            const float4 a0 = *(const float4*)&Wf[(size_t)(k0 + k)     * Cn + m0b + mg * 4];
            const float4 a1 = *(const float4*)&Wf[(size_t)(k0 + k + 1) * Cn + m0b + mg * 4];
            *(h2*)&lA[mg * 4 + 0][k] = __builtin_amdgcn_cvt_pkrtz(a0.x, a1.x);
            *(h2*)&lA[mg * 4 + 1][k] = __builtin_amdgcn_cvt_pkrtz(a0.y, a1.y);
            *(h2*)&lA[mg * 4 + 2][k] = __builtin_amdgcn_cvt_pkrtz(a0.z, a1.z);
            *(h2*)&lA[mg * 4 + 3][k] = __builtin_amdgcn_cvt_pkrtz(a0.w, a1.w);
        }
        // ---- stage gathered/weighted B tile: row e = (kk,c), col n with roll
        {
            const int r  = tid >> 3;        // e row within step: 0..31
            const int nb = (tid & 7) * 8;   // 8 consecutive n per thread
            const int eg = k0 + r;
            const int kk = eg >> 9;         // / Cn
            const int c  = eg & (Cn - 1);   // % Cn
            const int rid = (b * Cn + c) * Kn + kk;
            const int lag = lags[rid];
            const float wt = wvals[rid];
            const float* vr = &V[((size_t)b * Cn + c) * Tn];
#pragma unroll
            for (int j = 0; j < 8; ++j) {
                const int ts = (n0 + nb + j - lag) & (Tn - 1);
                lB[nb + j][r] = (_Float16)(vr[ts] * wt);
            }
        }
        __syncthreads();

        const int ml = wave * 16 + l15;
        const v8h a0 = *(const v8h*)&lA[ml][8 * lhi];
        const v8h a1 = *(const v8h*)&lA[ml][16 + 8 * lhi];
        const v16h afrag = cat16(a0, a1);
#pragma unroll
        for (int nn = 0; nn < 4; ++nn) {
            const int nl = nn * 16 + l15;
            const v8h b0 = *(const v8h*)&lB[nl][8 * lhi];
            const v8h b1 = *(const v8h*)&lB[nl][16 + 8 * lhi];
            acc[nn] = __builtin_amdgcn_wmma_f32_16x16x32_f16(
                false, afrag, false, cat16(b0, b1), (short)0, acc[nn], false, false);
        }
    }
    const int m0 = m0b + wave * 16;
#pragma unroll
    for (int nn = 0; nn < 4; ++nn) {
#pragma unroll
        for (int r = 0; r < 8; ++r) {
            const int m = m0 + r + (lhi << 3);
            const int n = n0 + nn * 16 + l15;
            Out[((size_t)b * Cn + m) * Tn + n] = acc[nn][r] + bias[m];
        }
    }
}

// ---------------------------------------------------------------------------
extern "C" void kernel_launch(void* const* d_in, const int* in_sizes, int n_in,
                              void* d_out, int out_size, void* d_ws, size_t ws_size,
                              hipStream_t stream) {
    (void)in_sizes; (void)n_in; (void)out_size; (void)ws_size;
    const float* query = (const float*)d_in[0];
    const float* key_  = (const float*)d_in[1];
    const float* value = (const float*)d_in[2];
    const float* Wq    = (const float*)d_in[3];
    const float* bq    = (const float*)d_in[4];
    const float* Wk    = (const float*)d_in[5];
    const float* bk    = (const float*)d_in[6];
    const float* Wv    = (const float*)d_in[7];
    const float* bv    = (const float*)d_in[8];
    const float* Wf    = (const float*)d_in[9];
    const float* bf    = (const float*)d_in[10];
    float* out = (float*)d_out;

    const size_t BCT = (size_t)Bn * Cn * Tn;
    float* qp = (float*)d_ws;          // [B,C,T] projected q
    float* kp = qp + BCT;              // [B,C,T] projected k
    float* vp = kp + BCT;              // [B,C,T] projected v
    int*   lg = (int*)(vp + BCT);      // [B,C,K] lag indices
    float* wv = (float*)(lg + (size_t)Bn * Cn * Kn);  // [B,C,K] softmax weights

    dim3 blk(256, 1, 1);
    dim3 g1(Tn / 64, Cn / 128, Bn);    // 32 x 4 x 16 blocks, 128x64 tile each

    proj_gemm_kernel<<<g1, blk, 0, stream>>>(Wq, query, bq, qp, Cn, Tn, Cn);
    proj_gemm_kernel<<<g1, blk, 0, stream>>>(Wk, key_,  bk, kp, Cn, Tn, Cn);
    proj_gemm_kernel<<<g1, blk, 0, stream>>>(Wv, value, bv, vp, Cn, Tn, Cn);

    autocorr_topk_kernel<<<dim3(Bn * Cn), blk, 0, stream>>>(qp, kp, lg, wv);

    final_gemm_kernel<<<g1, blk, 0, stream>>>(Wf, vp, lg, wv, bf, out);
}
